// TransformerLayer_65137474011256
// MI455X (gfx1250) — matrix operand order repaired and checked by reading, not compiled, not run on hardware
//
#include <hip/hip_runtime.h>
#include <hip/hip_bf16.h>

// ---------------------------------------------------------------------------
// CDNA5 (gfx1250) implementation: wave32, WMMA bf16 hi/lo split for ~fp32
// accuracy, flash-attention with async global->LDS tile staging (ASYNCcnt).
// ---------------------------------------------------------------------------

typedef __attribute__((ext_vector_type(16))) __bf16 v16bf;
typedef __attribute__((ext_vector_type(8)))  float  v8f;

#define WMMA_BF16(A, B, C) \
  __builtin_amdgcn_wmma_f32_16x16x32_bf16(false, (A), false, (B), (short)0, (C), false, false)

__device__ __forceinline__ v8f v8f_zero() {
  v8f r;
#pragma unroll
  for (int i = 0; i < 8; ++i) r[i] = 0.0f;
  return r;
}

__device__ __forceinline__ void split_bf16(float x, __bf16& hi, __bf16& lo) {
  hi = (__bf16)x;
  lo = (__bf16)(x - (float)hi);
}

// LDS byte offset (workgroup-relative) from a __shared__ pointer: the
// local->generic addrspacecast keeps the LDS offset in addr[31:0].
__device__ __forceinline__ unsigned lds_off(const void* p) {
  return (unsigned)(unsigned long long)p;
}

// Async copy: 16 bytes per lane, global (SGPR base + VGPR offset) -> LDS.
__device__ __forceinline__ void async_b128(unsigned lds, unsigned goff, const void* sbase) {
  asm volatile("global_load_async_to_lds_b128 %0, %1, %2"
               :: "v"(lds), "v"(goff), "s"(sbase) : "memory");
}
// Async copy: 4 bytes per lane (used for the transposed V staging).
__device__ __forceinline__ void async_b32(unsigned lds, unsigned goff, const void* sbase) {
  asm volatile("global_load_async_to_lds_b32 %0, %1, %2"
               :: "v"(lds), "v"(goff), "s"(sbase) : "memory");
}
__device__ __forceinline__ void wait_async0() {
  asm volatile("s_wait_asynccnt 0x0" ::: "memory");
}

// ---------------------------------------------------------------------------
// GEMM:  Y[M,Nc] = ([X1 | X2] @ W + bias) * post_scale (+ res)
// M = 8192 always. K1,K2 multiples of 32 (K2 may be 0). Nc multiple of 64.
// 64x64 output tile per 256-thread workgroup; 8 waves x two 16x16 WMMA tiles.
// bf16 hi/lo tiles staged in LDS (converted once, reused by all 8 waves).
// ---------------------------------------------------------------------------
__global__ __launch_bounds__(256) void gemm_bias_kernel(
    const float* __restrict__ X1, const float* __restrict__ X2,
    const float* __restrict__ W, const float* __restrict__ bias,
    const float* __restrict__ res, float* __restrict__ Y,
    int K1, int K2, int Nc, float post_scale)
{
  const int n0 = blockIdx.x * 64;
  const int m0 = blockIdx.y * 64;
  const int tid = threadIdx.x;
  const int wave = tid >> 5, lane = tid & 31;
  const int half = lane >> 4, ln = lane & 15;

  __shared__ __bf16 Ah[64][36];
  __shared__ __bf16 Al[64][36];
  __shared__ __bf16 Bh[64][36];   // stored transposed: [n][k]
  __shared__ __bf16 Bl[64][36];

  const int t0 = 2 * wave;
  const int tr0 = t0 & 3, tr1 = (t0 + 1) & 3, tc = t0 >> 2;

  v8f acc0 = v8f_zero(), acc1 = v8f_zero();

  const int K = K1 + K2;
  const int arow = tid >> 2, acg = (tid & 3) * 8;
  const int bkr = tid >> 3, bng = (tid & 7) * 8;

  for (int kb = 0; kb < K; kb += 32) {
    const float* Xs; int Ks, kk;
    if (kb < K1) { Xs = X1; Ks = K1; kk = kb; }
    else         { Xs = X2; Ks = K2; kk = kb - K1; }

    {
      const float* src = Xs + (size_t)(m0 + arow) * Ks + kk + acg;
#pragma unroll
      for (int i = 0; i < 8; ++i) {
        __bf16 hi, lo; split_bf16(src[i], hi, lo);
        Ah[arow][acg + i] = hi; Al[arow][acg + i] = lo;
      }
    }
    {
      const float* src = W + (size_t)(kb + bkr) * Nc + n0 + bng;
#pragma unroll
      for (int i = 0; i < 8; ++i) {
        __bf16 hi, lo; split_bf16(src[i], hi, lo);
        Bh[bng + i][bkr] = hi; Bl[bng + i][bkr] = lo;
      }
    }
    if (kb + 32 < K) {
      __builtin_prefetch(W + (size_t)(kb + 32 + bkr) * Nc + n0 + bng, 0, 1);
    }
    __syncthreads();

    // B fragment (32x16): lane n = tc*16+ln; K = (e<8) ? 8*half+e : 8+8*half+e
    v16bf bfh, bfl;
    const int nl = tc * 16 + ln;
#pragma unroll
    for (int e = 0; e < 16; e += 2) {
      const int Kk = (e < 8) ? (8 * half + e) : (8 + 8 * half + e);
      bfh[e] = Bh[nl][Kk]; bfh[e + 1] = Bh[nl][Kk + 1];
      bfl[e] = Bl[nl][Kk]; bfl[e + 1] = Bl[nl][Kk + 1];
    }
    // A fragments (16x32): lane m; K = (e<8) ? 8*half+e : 8+8*half+e
    v16bf ah0, al0, ah1, al1;
    const int ml0 = tr0 * 16 + ln, ml1 = tr1 * 16 + ln;
#pragma unroll
    for (int e = 0; e < 16; e += 2) {
      const int Ko = (e < 8) ? (8 * half + e) : (8 + 8 * half + e);
      ah0[e] = Ah[ml0][Ko]; ah0[e + 1] = Ah[ml0][Ko + 1];
      al0[e] = Al[ml0][Ko]; al0[e + 1] = Al[ml0][Ko + 1];
      ah1[e] = Ah[ml1][Ko]; ah1[e + 1] = Ah[ml1][Ko + 1];
      al1[e] = Al[ml1][Ko]; al1[e + 1] = Al[ml1][Ko + 1];
    }
    acc0 = WMMA_BF16(ah0, bfh, acc0);
    acc0 = WMMA_BF16(ah0, bfl, acc0);
    acc0 = WMMA_BF16(al0, bfh, acc0);
    acc1 = WMMA_BF16(ah1, bfh, acc1);
    acc1 = WMMA_BF16(ah1, bfl, acc1);
    acc1 = WMMA_BF16(al1, bfh, acc1);
    __syncthreads();
  }

  const int nc = n0 + tc * 16 + ln;
  const float bv = bias[nc];
#pragma unroll
  for (int r = 0; r < 8; ++r) {
    const int m = m0 + tr0 * 16 + r + 8 * half;
    float v = (acc0[r] + bv) * post_scale;
    if (res) v += res[(size_t)m * Nc + nc];
    Y[(size_t)m * Nc + nc] = v;
  }
#pragma unroll
  for (int r = 0; r < 8; ++r) {
    const int m = m0 + tr1 * 16 + r + 8 * half;
    float v = (acc1[r] + bv) * post_scale;
    if (res) v += res[(size_t)m * Nc + nc];
    Y[(size_t)m * Nc + nc] = v;
  }
}

// ---------------------------------------------------------------------------
// RoPE + qkv de-interleave. qkv layout per reference reshape(B,N,H,HD,3):
// qkv[bn*768 + h*192 + hd*3 + c]. Writes Q/K/V as [B,N,H*HD] fp32.
// ---------------------------------------------------------------------------
__global__ __launch_bounds__(256) void rope_split_kernel(
    const float* __restrict__ qkv, const float* __restrict__ enc,
    float* __restrict__ Qb, float* __restrict__ Kb, float* __restrict__ Vb)
{
  const size_t g = (size_t)blockIdx.x * 256 + threadIdx.x;  // [0, B*N*H*32)
  const int i = (int)(g & 31);
  const size_t t = g >> 5;
  const int h = (int)(t & 3);
  const size_t bn = t >> 2;  // [0, 8192)

  const float* src = qkv + bn * 768 + (size_t)h * 192 + (size_t)(2 * i) * 3;
  const float qe = src[0], ke = src[1], ve = src[2];
  const float qo = src[3], ko = src[4], vo = src[5];

  const float* e0 = enc + bn * 64 + 2 * i;
  const float* e1 = enc + (size_t)524288 + bn * 64 + 2 * i;
  const float f0e = e0[0], f0o = e0[1], f1e = e1[0], f1o = e1[1];

  const size_t dst = bn * 256 + (size_t)h * 64 + 2 * i;
  Qb[dst]     = qe * f0e - qo * f1e;
  Qb[dst + 1] = qo * f0o + qe * f1o;
  Kb[dst]     = ke * f0e - ko * f1e;
  Kb[dst + 1] = ko * f0o + ke * f1o;
  Vb[dst]     = ve;
  Vb[dst + 1] = vo;
}

// ---------------------------------------------------------------------------
// Flash attention. Q,K,V,O in [B,N,H*HD]=[B,2048,256] layout; head h at col h*64.
// Grid (B*H, N/64); 128 threads = 4 waves; wave owns 16 query rows.
// K/V tiles staged via async global->LDS (f32); bf16 hi/lo split happens at
// fragment build (VALU co-executes with XDL WMMA).
// ---------------------------------------------------------------------------
__global__ __launch_bounds__(128) void flash_attn_kernel(
    const float* __restrict__ Q, const float* __restrict__ K,
    const float* __restrict__ V, float* __restrict__ O, float qscale)
{
  const int bh = blockIdx.x;
  const int b = bh >> 2, h = bh & 3;
  const int m0 = blockIdx.y * 64;
  const size_t base = ((size_t)b * 2048) * 256 + (size_t)h * 64;
  const int tid = threadIdx.x, wave = tid >> 5, lane = tid & 31;
  const int half = lane >> 4, ln = lane & 15;

  __shared__ float  Kf[64][68];   // [j][hd]; 272B row stride (16B multiple)
  __shared__ float  Vt[64][68];   // transposed: [hd][j]
  __shared__ __bf16 Ph[64][66];   // [m][j], per-wave private 16-row strips
  __shared__ __bf16 Pl[64][66];

  // Load Q A-fragments once (2 chunks of K=32 over hd=64), scale folded in.
  const float qs = qscale * 1.4426950408889634f;  // * log2(e)
  const int mq = m0 + wave * 16 + ln;
  v16bf qh[2], ql[2];
#pragma unroll
  for (int kc = 0; kc < 2; ++kc) {
#pragma unroll
    for (int e = 0; e < 16; e += 2) {
      const int Ko = (e < 8) ? (8 * half + e) : (8 + 8 * half + e);
      const int hd = kc * 32 + Ko;
      const float x0 = Q[base + (size_t)mq * 256 + hd] * qs;
      const float x1 = Q[base + (size_t)mq * 256 + hd + 1] * qs;
      __bf16 h0, l0, h1, l1;
      split_bf16(x0, h0, l0); split_bf16(x1, h1, l1);
      qh[kc][e] = h0; qh[kc][e + 1] = h1;
      ql[kc][e] = l0; ql[kc][e + 1] = l1;
    }
  }

  const float* kblk = K + base;  // + j*256 + hd
  const float* vblk = V + base;

  v8f o[4];
  float rm[8], rl[8];
#pragma unroll
  for (int t = 0; t < 4; ++t) o[t] = v8f_zero();
#pragma unroll
  for (int r = 0; r < 8; ++r) { rm[r] = -1.0e30f; rl[r] = 0.0f; }

  for (int j0 = 0; j0 < 2048; j0 += 64) {
    __syncthreads();  // previous iteration's LDS reads complete

    // --- async stage K tile: 64 rows x 256B, 16B chunks (1024 lane-ops) ---
    {
      const unsigned ldsK = lds_off(&Kf[0][0]);
#pragma unroll
      for (int i = 0; i < 8; ++i) {
        const int c = i * 128 + tid;          // chunk id
        const int row = c >> 4, ch = c & 15;  // 16B chunk within row
        const unsigned goff = (unsigned)((j0 + row) * 1024 + ch * 16);
        async_b128(ldsK + (unsigned)(row * 272 + ch * 16), goff, (const void*)kblk);
      }
    }
    // --- async stage V tile transposed: per-element b32 (4096 lane-ops) ---
    {
      const unsigned ldsV = lds_off(&Vt[0][0]);
#pragma unroll
      for (int i = 0; i < 32; ++i) {
        const int e = i * 128 + tid;
        const int row = e >> 6, hd = e & 63;
        const unsigned goff = (unsigned)((j0 + row) * 1024 + hd * 4);
        async_b32(ldsV + (unsigned)(hd * 272 + row * 4), goff, (const void*)vblk);
      }
    }
    wait_async0();
    __syncthreads();  // tiles visible to all waves

    // S = Q @ K^T : 4 col-tiles of 16, K-dim = hd (2 chunks of 32).
    v8f st[4];
#pragma unroll
    for (int jt = 0; jt < 4; ++jt) {
      st[jt] = v8f_zero();
      const int nl = jt * 16 + ln;
#pragma unroll
      for (int kc = 0; kc < 2; ++kc) {
        v16bf kbh, kbl;
#pragma unroll
        for (int e = 0; e < 16; e += 2) {
          const int Kk = kc * 32 + ((e < 8) ? (8 * half + e) : (8 + 8 * half + e));
          __bf16 h0, l0, h1, l1;
          split_bf16(Kf[nl][Kk], h0, l0);
          split_bf16(Kf[nl][Kk + 1], h1, l1);
          kbh[e] = h0; kbh[e + 1] = h1;
          kbl[e] = l0; kbl[e + 1] = l1;
        }
        st[jt] = WMMA_BF16(qh[kc], kbh, st[jt]);
        st[jt] = WMMA_BF16(qh[kc], kbl, st[jt]);
        st[jt] = WMMA_BF16(ql[kc], kbh, st[jt]);
      }
    }

    // Online softmax. C-layout: row = r + 8*half, col = jt*16 + ln.
#pragma unroll
    for (int r = 0; r < 8; ++r) {
      float mx = fmaxf(fmaxf(st[0][r], st[1][r]), fmaxf(st[2][r], st[3][r]));
#pragma unroll
      for (int msk = 1; msk < 16; msk <<= 1) mx = fmaxf(mx, __shfl_xor(mx, msk, 32));
      const float nm = fmaxf(rm[r], mx);
      const float corr = exp2f(rm[r] - nm);
      rm[r] = nm;
      float sum = 0.0f;
#pragma unroll
      for (int jt = 0; jt < 4; ++jt) {
        const float p = exp2f(st[jt][r] - nm);
        st[jt][r] = p;
        sum += p;
      }
#pragma unroll
      for (int msk = 1; msk < 16; msk <<= 1) sum += __shfl_xor(sum, msk, 32);
      rl[r] = rl[r] * corr + sum;
#pragma unroll
      for (int ht = 0; ht < 4; ++ht) o[ht][r] *= corr;
    }

    // Write P (C-layout) to this wave's private LDS strip; reload as A-frag.
    const int prow0 = wave * 16 + 8 * half;
#pragma unroll
    for (int jt = 0; jt < 4; ++jt) {
      const int pc = jt * 16 + ln;
#pragma unroll
      for (int r = 0; r < 8; ++r) {
        __bf16 hi, lo; split_bf16(st[jt][r], hi, lo);
        Ph[prow0 + r][pc] = hi;
        Pl[prow0 + r][pc] = lo;
      }
    }

    // O += P @ V : 4 hd-tiles, K-dim = j (2 chunks of 32).
    const int ml = wave * 16 + ln;
#pragma unroll
    for (int kc = 0; kc < 2; ++kc) {
      v16bf pah, pal;
#pragma unroll
      for (int e = 0; e < 16; e += 2) {
        const int Ko = (e < 8) ? (8 * half + e) : (8 + 8 * half + e);
        const int j = kc * 32 + Ko;
        pah[e] = Ph[ml][j]; pah[e + 1] = Ph[ml][j + 1];
        pal[e] = Pl[ml][j]; pal[e + 1] = Pl[ml][j + 1];
      }
#pragma unroll
      for (int ht = 0; ht < 4; ++ht) {
        v16bf vbh, vbl;
        const int nl = ht * 16 + ln;
#pragma unroll
        for (int e = 0; e < 16; e += 2) {
          const int Kk = kc * 32 + ((e < 8) ? (8 * half + e) : (8 + 8 * half + e));
          __bf16 h0, l0, h1, l1;
          split_bf16(Vt[nl][Kk], h0, l0);
          split_bf16(Vt[nl][Kk + 1], h1, l1);
          vbh[e] = h0; vbh[e + 1] = h1;
          vbl[e] = l0; vbl[e + 1] = l1;
        }
        o[ht] = WMMA_BF16(pah, vbh, o[ht]);
        o[ht] = WMMA_BF16(pah, vbl, o[ht]);
        o[ht] = WMMA_BF16(pal, vbh, o[ht]);
      }
    }
  }

#pragma unroll
  for (int r = 0; r < 8; ++r) {
    const float inv = 1.0f / rl[r];
    const size_t mg = (size_t)(m0 + wave * 16 + r + 8 * half);
#pragma unroll
    for (int ht = 0; ht < 4; ++ht) {
      O[base + mg * 256 + ht * 16 + ln] = o[ht][r] * inv;
    }
  }
}

// ---------------------------------------------------------------------------
// Row LayerNorm (population variance) + exact GELU, in place. Rows of 512.
// ---------------------------------------------------------------------------
__global__ __launch_bounds__(256) void ln_gelu_kernel(
    float* __restrict__ X, const float* __restrict__ g, const float* __restrict__ be)
{
  const int row = blockIdx.x;
  float* xr = X + (size_t)row * 512;
  const int tid = threadIdx.x;
  const float x0 = xr[tid], x1 = xr[tid + 256];
  float s = x0 + x1, s2 = x0 * x0 + x1 * x1;
#pragma unroll
  for (int msk = 1; msk < 32; msk <<= 1) {
    s += __shfl_xor(s, msk, 32);
    s2 += __shfl_xor(s2, msk, 32);
  }
  __shared__ float bs[8], bs2[8];
  const int wave = tid >> 5, lane = tid & 31;
  if (lane == 0) { bs[wave] = s; bs2[wave] = s2; }
  __syncthreads();
  float ts = 0.0f, ts2 = 0.0f;
#pragma unroll
  for (int i = 0; i < 8; ++i) { ts += bs[i]; ts2 += bs2[i]; }
  const float mean = ts * (1.0f / 512.0f);
  const float var = ts2 * (1.0f / 512.0f) - mean * mean;
  const float inv = rsqrtf(var + 1e-5f);

  float y0 = (x0 - mean) * inv * g[tid] + be[tid];
  float y1 = (x1 - mean) * inv * g[tid + 256] + be[tid + 256];
  xr[tid]       = 0.5f * y0 * (1.0f + erff(y0 * 0.7071067811865476f));
  xr[tid + 256] = 0.5f * y1 * (1.0f + erff(y1 * 0.7071067811865476f));
}

// ---------------------------------------------------------------------------
// Host launcher
// ---------------------------------------------------------------------------
extern "C" void kernel_launch(void* const* d_in, const int* in_sizes, int n_in,
                              void* d_out, int out_size, void* d_ws, size_t ws_size,
                              hipStream_t stream) {
  (void)in_sizes; (void)n_in; (void)out_size; (void)ws_size;

  const float* desc0 = (const float*)d_in[0];
  const float* desc1 = (const float*)d_in[1];
  const float* enc0  = (const float*)d_in[2];
  const float* enc1  = (const float*)d_in[3];
  const float* Wqkv  = (const float*)d_in[4];
  const float* bqkv  = (const float*)d_in[5];
  const float* Wo_s  = (const float*)d_in[6];
  const float* bo_s  = (const float*)d_in[7];
  const float* W1_s  = (const float*)d_in[8];
  const float* b1_s  = (const float*)d_in[9];
  const float* g_s   = (const float*)d_in[10];
  const float* be_s  = (const float*)d_in[11];
  const float* W2_s  = (const float*)d_in[12];
  const float* b2_s  = (const float*)d_in[13];
  const float* Wqk_c = (const float*)d_in[14];
  const float* bqk_c = (const float*)d_in[15];
  const float* Wv_c  = (const float*)d_in[16];
  const float* bv_c  = (const float*)d_in[17];
  const float* Wo_c  = (const float*)d_in[18];
  const float* bo_c  = (const float*)d_in[19];
  const float* W1_c  = (const float*)d_in[20];
  const float* b1_c  = (const float*)d_in[21];
  const float* g_c   = (const float*)d_in[22];
  const float* be_c  = (const float*)d_in[23];
  const float* W2_c  = (const float*)d_in[24];
  const float* b2_c  = (const float*)d_in[25];

  const size_t R = 8192;  // B*N
  float* ws   = (float*)d_ws;
  float* qkvb = ws;                 // R*768
  float* Qb   = qkvb + R * 768;     // R*256 each below
  float* Kb   = Qb + R * 256;
  float* Vb   = Kb + R * 256;
  float* Ob   = Vb + R * 256;
  float* Msg  = Ob + R * 256;
  float* Hb   = Msg + R * 256;      // R*512
  float* SX0  = Hb + R * 512;
  float* SX1  = SX0 + R * 256;
  float* M0r  = qkvb;               // reuse qkv area in cross phase
  float* M1r  = qkvb + R * 256;

  float* out0 = (float*)d_out;
  float* out1 = out0 + R * 256;

  auto gemm = [&](const float* X1, const float* X2, const float* W, const float* bvec,
                  const float* resid, float* Y, int K1, int K2, int Nc, float scale) {
    dim3 grid(Nc / 64, 128);
    gemm_bias_kernel<<<grid, 256, 0, stream>>>(X1, X2, W, bvec, resid, Y, K1, K2, Nc, scale);
  };

  auto self_block = [&](const float* x, const float* enc, float* SX) {
    gemm(x, nullptr, Wqkv, bqkv, nullptr, qkvb, 256, 0, 768, 1.0f);
    rope_split_kernel<<<4096, 256, 0, stream>>>(qkvb, enc, Qb, Kb, Vb);
    flash_attn_kernel<<<dim3(16, 32), 128, 0, stream>>>(Qb, Kb, Vb, Ob, 0.125f);
    gemm(Ob, nullptr, Wo_s, bo_s, nullptr, Msg, 256, 0, 256, 1.0f);
    gemm(x, Msg, W1_s, b1_s, nullptr, Hb, 256, 256, 512, 1.0f);
    ln_gelu_kernel<<<8192, 256, 0, stream>>>(Hb, g_s, be_s);
    gemm(Hb, nullptr, W2_s, b2_s, x, SX, 512, 0, 256, 1.0f);
  };

  // Self blocks
  self_block(desc0, enc0, SX0);
  self_block(desc1, enc1, SX1);

  // Cross block. scale s = HD^-0.5 = 0.125; each qk scaled by sqrt(s).
  const float sq = 0.35355339059327373f;
  gemm(SX0, nullptr, Wqk_c, bqk_c, nullptr, Qb, 256, 0, 256, sq);   // qk0
  gemm(SX1, nullptr, Wqk_c, bqk_c, nullptr, Kb, 256, 0, 256, sq);   // qk1
  gemm(SX0, nullptr, Wv_c, bv_c, nullptr, Vb, 256, 0, 256, 1.0f);   // v0
  gemm(SX1, nullptr, Wv_c, bv_c, nullptr, Ob, 256, 0, 256, 1.0f);   // v1

  // m0 = softmax(qk0·qk1^T) @ v1 ; m1 = softmax(qk1·qk0^T) @ v0
  flash_attn_kernel<<<dim3(16, 32), 128, 0, stream>>>(Qb, Kb, Ob, M0r, 1.0f);
  flash_attn_kernel<<<dim3(16, 32), 128, 0, stream>>>(Kb, Qb, Vb, M1r, 1.0f);

  // x0 branch
  gemm(M0r, nullptr, Wo_c, bo_c, nullptr, Msg, 256, 0, 256, 1.0f);
  gemm(SX0, Msg, W1_c, b1_c, nullptr, Hb, 256, 256, 512, 1.0f);
  ln_gelu_kernel<<<8192, 256, 0, stream>>>(Hb, g_c, be_c);
  gemm(Hb, nullptr, W2_c, b2_c, SX0, out0, 512, 0, 256, 1.0f);

  // x1 branch
  gemm(M1r, nullptr, Wo_c, bo_c, nullptr, Msg, 256, 0, 256, 1.0f);
  gemm(SX1, Msg, W1_c, b1_c, nullptr, Hb, 256, 256, 512, 1.0f);
  ln_gelu_kernel<<<8192, 256, 0, stream>>>(Hb, g_c, be_c);
  gemm(Hb, nullptr, W2_c, b2_c, SX1, out1, 512, 0, 256, 1.0f);
}
